// ConViT_41506563948681
// MI455X (gfx1250) — matrix operand order repaired and checked by reading, not compiled, not run on hardware
//
#include <hip/hip_runtime.h>
#include <hip/hip_bf16.h>

// ---------------------------------------------------------------------------
// ConViT forward for MI455X (gfx1250, wave32, WMMA).
// bf16 WMMA GEMM with fp32 accumulate, vectorized fp32->bf16 LDS staging,
// block-uniform fast path, double-buffered LDS, 4 WMMAs per wave per K-step.
// ---------------------------------------------------------------------------

typedef __attribute__((ext_vector_type(16))) __bf16 v16bf;
typedef __attribute__((ext_vector_type(8)))  __bf16 v8bf;
typedef __attribute__((ext_vector_type(8)))  float  v8f;

#define Bsz   64
#define NH    12
#define Dh    64
#define Cdim  768
#define C2    1536
#define C3    2304
#define C4    3072
#define N196  196
#define N197  197
#define GRID_ 14
#define NCLS  1000
#define SCALE_ 0.125f
#define SPAD  208      // padded score-row stride (16B-aligned float rows)

#define TILE 64
#define BK   32
#define LDSW (BK + 8)  // 40 bf16 = 80B row; 80 % 16 == 0 keeps 16B stores aligned

__device__ __forceinline__ v8bf pack8(float4 a, float4 b) {
  v8bf r;
  r[0] = (__bf16)a.x; r[1] = (__bf16)a.y; r[2] = (__bf16)a.z; r[3] = (__bf16)a.w;
  r[4] = (__bf16)b.x; r[5] = (__bf16)b.y; r[6] = (__bf16)b.z; r[7] = (__bf16)b.w;
  return r;
}

// ---------------------------------------------------------------------------
// Generic batched WMMA GEMM (64x64 tile, 128 threads = 4 waves, each wave
// computes a 16x64 strip = 4 accumulators; double-buffered LDS):
//   C[m,n] = act( alpha * sum_k A[m,k] * B'[n,k] + bias[n] ) + resid[m,n]
// B' = Bw (NT, rows of length K) when bT==0; B' = Bw^T (Bw is KxNout) when bT==1.
// ---------------------------------------------------------------------------
__global__ __launch_bounds__(128) void gemm_wmma_kernel(
    const float* __restrict__ A, const float* __restrict__ Bw,
    float* __restrict__ Cc, const float* __restrict__ bias,
    const float* __restrict__ resid,
    int M, int Nout, int K, int lda, int ldb, int ldc,
    long long a_so, long long a_si, long long b_so, long long b_si,
    long long c_so, long long c_si, int batch_h,
    float alpha, int bT, int act)
{
  const int z  = blockIdx.z;
  const long long zo = z / batch_h, zi = z % batch_h;
  const float* Ab = A  + zo * a_so + zi * a_si;
  const float* Bb = Bw + zo * b_so + zi * b_si;
  float*       Cb = Cc + zo * c_so + zi * c_si;
  const float* Rb = resid ? (resid + zo * c_so + zi * c_si) : nullptr;

  __shared__ __bf16 As[2][TILE][LDSW];
  __shared__ __bf16 Bs[2][TILE][LDSW];

  const int t    = threadIdx.x;
  const int wave = t >> 5, lane = t & 31;
  const int lr   = lane & 15, lh = lane >> 4;
  const int m0   = blockIdx.y * TILE;
  const int n0   = blockIdx.x * TILE;

  // staging coordinates (hoisted out of the K loop)
  const int rA = t >> 1;            // 0..63 : A/B row (NT)
  const int cA = (t & 1) * 16;      // 0/16  : k group
  const int kB = t >> 2;            // 0..31 : k row (bT)
  const int nB = (t & 3) * 16;      // n group (bT)
  const float* aPtr  = Ab + (long long)(m0 + rA) * lda + cA;
  const float* bPtrN = Bb + (long long)(n0 + rA) * ldb + cA;
  const float* bPtrT = Bb + (long long)kB * ldb + n0 + nB;

  // block-uniform edge classification: interior blocks take a guard-free path
  const bool edge = (m0 + TILE > M) || (n0 + TILE > Nout);

  auto stage = [&](int kb, int pp) {
    if (!edge && kb + BK <= K) {
      // ---- fast path: no per-thread guards, vector loads, 16B LDS stores ----
      {
        const float4* p = (const float4*)(aPtr + kb);
        float4 x0 = p[0], x1 = p[1], x2 = p[2], x3 = p[3];
        *(v8bf*)&As[pp][rA][cA]     = pack8(x0, x1);
        *(v8bf*)&As[pp][rA][cA + 8] = pack8(x2, x3);
      }
      if (!bT) {
        const float4* p = (const float4*)(bPtrN + kb);
        float4 x0 = p[0], x1 = p[1], x2 = p[2], x3 = p[3];
        *(v8bf*)&Bs[pp][rA][cA]     = pack8(x0, x1);
        *(v8bf*)&Bs[pp][rA][cA + 8] = pack8(x2, x3);
      } else {
        const float4* p = (const float4*)(bPtrT + (long long)kb * ldb);
        float4 x0 = p[0], x1 = p[1], x2 = p[2], x3 = p[3];
        Bs[pp][nB +  0][kB] = (__bf16)x0.x; Bs[pp][nB +  1][kB] = (__bf16)x0.y;
        Bs[pp][nB +  2][kB] = (__bf16)x0.z; Bs[pp][nB +  3][kB] = (__bf16)x0.w;
        Bs[pp][nB +  4][kB] = (__bf16)x1.x; Bs[pp][nB +  5][kB] = (__bf16)x1.y;
        Bs[pp][nB +  6][kB] = (__bf16)x1.z; Bs[pp][nB +  7][kB] = (__bf16)x1.w;
        Bs[pp][nB +  8][kB] = (__bf16)x2.x; Bs[pp][nB +  9][kB] = (__bf16)x2.y;
        Bs[pp][nB + 10][kB] = (__bf16)x2.z; Bs[pp][nB + 11][kB] = (__bf16)x2.w;
        Bs[pp][nB + 12][kB] = (__bf16)x3.x; Bs[pp][nB + 13][kB] = (__bf16)x3.y;
        Bs[pp][nB + 14][kB] = (__bf16)x3.z; Bs[pp][nB + 15][kB] = (__bf16)x3.w;
      }
    } else {
      // ---- guarded path: tile borders / K tails only ----
      const int gm = m0 + rA;
      #pragma unroll
      for (int i = 0; i < 16; ++i) {
        const int gk = kb + cA + i;
        float v = (gm < M && gk < K) ? Ab[(long long)gm * lda + gk] : 0.f;
        As[pp][rA][cA + i] = (__bf16)v;
      }
      if (!bT) {
        const int gn = n0 + rA;
        #pragma unroll
        for (int i = 0; i < 16; ++i) {
          const int gk = kb + cA + i;
          float v = (gn < Nout && gk < K) ? Bb[(long long)gn * ldb + gk] : 0.f;
          Bs[pp][rA][cA + i] = (__bf16)v;
        }
      } else {
        const int gk = kb + kB;
        #pragma unroll
        for (int j = 0; j < 16; ++j) {
          const int gn = n0 + nB + j;
          float v = (gk < K && gn < Nout) ? Bb[(long long)gk * ldb + gn] : 0.f;
          Bs[pp][nB + j][kB] = (__bf16)v;
        }
      }
    }
  };

  v8f acc[4] = {v8f{}, v8f{}, v8f{}, v8f{}};

  stage(0, 0);
  __syncthreads();

  int p = 0;
  for (int kb = 0; kb < K; kb += BK) {
    // prefetch two tiles ahead (gfx1250 global_prefetch_b8)
    if (kb + 2 * BK < K)
      __builtin_prefetch(aPtr + kb + 2 * BK, 0, 1);
    // stage next tile into the other buffer
    if (kb + BK < K) stage(kb + BK, p ^ 1);

    // A fragment (16x32): lanes 0-15 M=lane K=0..15; lanes 16-31 K=16..31
    v16bf af;
    {
      const int r  = wave * 16 + lr;
      const int k16 = lh * 16;
      #pragma unroll
      for (int i = 0; i < 16; ++i) {
        af[i]     = As[p][r][k16 + i];
      }
    }
    // 4 B fragments, 4 WMMAs (A fragment reused)
    #pragma unroll
    for (int j = 0; j < 4; ++j) {
      v16bf bf;
      const int rb  = j * 16 + lr;
      const int k16 = lh * 16;
      #pragma unroll
      for (int i = 0; i < 16; ++i) bf[i] = Bs[p][rb][k16 + i];
      acc[j] = __builtin_amdgcn_wmma_f32_16x16x32_bf16(false, af, false, bf,
                                                       (short)0, acc[j], false, false);
    }
    __syncthreads();
    p ^= 1;
  }

  // epilogue: D layout M = r + 8*lh, N = lr (per 16x16 tile)
  const int mrow = m0 + wave * 16 + lh * 8;
  #pragma unroll
  for (int j = 0; j < 4; ++j) {
    const int ncol = n0 + j * 16 + lr;
    if (ncol >= Nout) continue;
    const float bi = bias ? bias[ncol] : 0.f;
    #pragma unroll
    for (int r = 0; r < 8; ++r) {
      const int m = mrow + r;
      if (m >= M) continue;
      float v = alpha * acc[j][r] + bi;
      if (act) v = 0.5f * v * (1.f + erff(v * 0.70710678118654752f));
      if (Rb)  v += Rb[(long long)m * ldc + ncol];
      Cb[(long long)m * ldc + ncol] = v;
    }
  }
}

// ---------------------------------------------------------------------------
// LayerNorm over C=768 (3 elements / thread, 256 threads).
// ---------------------------------------------------------------------------
__global__ __launch_bounds__(256) void layernorm_kernel(
    const float* __restrict__ in, float* __restrict__ out,
    const float* __restrict__ s, const float* __restrict__ b,
    long long in_rs, long long out_rs)
{
  const long long row = blockIdx.x;
  const int t = threadIdx.x;
  const float* x = in + row * in_rs;
  float loc[3], sum = 0.f;
  #pragma unroll
  for (int i = 0; i < 3; ++i) { loc[i] = x[t + i * 256]; sum += loc[i]; }

  __shared__ float red[256];
  red[t] = sum; __syncthreads();
  for (int off = 128; off > 0; off >>= 1) {
    if (t < off) red[t] += red[t + off];
    __syncthreads();
  }
  const float mean = red[0] * (1.f / 768.f);
  __syncthreads();

  float vs = 0.f;
  #pragma unroll
  for (int i = 0; i < 3; ++i) { float d = loc[i] - mean; vs += d * d; }
  red[t] = vs; __syncthreads();
  for (int off = 128; off > 0; off >>= 1) {
    if (t < off) red[t] += red[t + off];
    __syncthreads();
  }
  const float inv = rsqrtf(red[0] * (1.f / 768.f) + 1e-5f);
  #pragma unroll
  for (int i = 0; i < 3; ++i) {
    const int c = t + i * 256;
    out[row * out_rs + c] = (loc[i] - mean) * inv * s[c] + b[c];
  }
}

// ---------------------------------------------------------------------------
// Row softmax, in place. L <= 256, one block per row, row stride ld.
// ---------------------------------------------------------------------------
__global__ __launch_bounds__(256) void softmax_kernel(
    float* __restrict__ data, int L, long long ld)
{
  const long long row = blockIdx.x;
  float* x = data + row * ld;
  const int t = threadIdx.x;
  float v = (t < L) ? x[t] : -3.4e38f;
  __shared__ float red[256];
  red[t] = v; __syncthreads();
  for (int off = 128; off > 0; off >>= 1) {
    if (t < off) red[t] = fmaxf(red[t], red[t + off]);
    __syncthreads();
  }
  const float mx = red[0]; __syncthreads();
  const float e = (t < L) ? expf(v - mx) : 0.f;
  red[t] = e; __syncthreads();
  for (int off = 128; off > 0; off >>= 1) {
    if (t < off) red[t] += red[t + off];
    __syncthreads();
  }
  if (t < L) x[t] = e / red[0];
}

// ---------------------------------------------------------------------------
// GPSA positional logits: pos[h,n,m] = pw[h,:] . (dx,dy,dx^2+dy^2) + pb[h]
// ---------------------------------------------------------------------------
__global__ void pos_logits_kernel(const float* __restrict__ pw,
                                  const float* __restrict__ pb,
                                  float* __restrict__ out, int total)
{
  const int idx = blockIdx.x * 256 + threadIdx.x;
  if (idx >= total) return;
  const int h = idx / (N196 * N196);
  const int r = idx % (N196 * N196);
  const int n = r / N196, m = r % N196;
  const float dx = (float)((m % GRID_) - (n % GRID_));
  const float dy = (float)((m / GRID_) - (n / GRID_));
  out[idx] = pw[h * 3 + 0] * dx + pw[h * 3 + 1] * dy +
             pw[h * 3 + 2] * (dx * dx + dy * dy) + pb[h];
}

// ---------------------------------------------------------------------------
// GPSA gate blend: attn = (1-g)*softmax(S) + g*pos_attn; renormalize.
// One block per (b,h,n) row; S rows have stride SPAD.
// ---------------------------------------------------------------------------
__global__ __launch_bounds__(256) void gpsa_blend_kernel(
    float* __restrict__ S, const float* __restrict__ posattn,
    const float* __restrict__ gate)
{
  const int rid = blockIdx.x;
  const int n = rid % N196;
  const int h = (rid / N196) % NH;
  const int t = threadIdx.x;
  const float g = 1.f / (1.f + expf(-gate[h]));

  float* x = S + (long long)rid * SPAD;
  const float* pp = posattn + ((long long)h * N196 + n) * N196;

  float v = (t < N196) ? x[t] : -3.4e38f;
  __shared__ float red[256];
  red[t] = v; __syncthreads();
  for (int off = 128; off > 0; off >>= 1) {
    if (t < off) red[t] = fmaxf(red[t], red[t + off]);
    __syncthreads();
  }
  const float mx = red[0]; __syncthreads();
  float e = (t < N196) ? expf(v - mx) : 0.f;
  red[t] = e; __syncthreads();
  for (int off = 128; off > 0; off >>= 1) {
    if (t < off) red[t] += red[t + off];
    __syncthreads();
  }
  const float inv = 1.f / red[0];
  __syncthreads();

  float blended = (t < N196) ? ((1.f - g) * (e * inv) + g * pp[t]) : 0.f;
  red[t] = blended; __syncthreads();
  for (int off = 128; off > 0; off >>= 1) {
    if (t < off) red[t] += red[t + off];
    __syncthreads();
  }
  if (t < N196) x[t] = blended / red[0];
}

// ---------------------------------------------------------------------------
// im2col for 16x16/stride-16 patches: out[b*196+p, c*256+py*16+px]
// ---------------------------------------------------------------------------
__global__ void im2col_kernel(const float* __restrict__ x, float* __restrict__ out,
                              int total)
{
  const int idx = blockIdx.x * 256 + threadIdx.x;
  if (idx >= total) return;
  const int m = idx / Cdim, k = idx % Cdim;
  const int b = m / N196, p = m % N196;
  const int c = k / 256, rr = k % 256;
  const int py = rr / 16, px = rr % 16;
  const int row = (p / GRID_) * 16 + py;
  const int col = (p % GRID_) * 16 + px;
  out[idx] = x[(((long long)b * 3 + c) * 224 + row) * 224 + col];
}

__global__ void add_pos_kernel(float* __restrict__ feat,
                               const float* __restrict__ pos, int total)
{
  const int idx = blockIdx.x * 256 + threadIdx.x;
  if (idx >= total) return;
  const int p = (idx / Cdim) % N196;
  const int c = idx % Cdim;
  feat[idx] += pos[p * Cdim + c];
}

__global__ void concat_cls_kernel(const float* __restrict__ in196,
                                  const float* __restrict__ cls,
                                  float* __restrict__ out197, int total)
{
  const int idx = blockIdx.x * 256 + threadIdx.x;
  if (idx >= total) return;
  const int m = idx / Cdim, c = idx % Cdim;
  const int b = m / N197, tkn = m % N197;
  out197[idx] = (tkn == 0) ? cls[c]
                           : in196[((long long)b * N196 + (tkn - 1)) * Cdim + c];
}

// ---------------------------------------------------------------------------
// Host-side orchestration
// ---------------------------------------------------------------------------
static void launch_gemm(hipStream_t st, const float* A, const float* Bw, float* Cc,
                        const float* bias, const float* resid,
                        int M, int Nout, int K, int lda, int ldb, int ldc,
                        long long a_so, long long a_si, long long b_so, long long b_si,
                        long long c_so, long long c_si, int batches, int batch_h,
                        float alpha, int bT, int act)
{
  dim3 grid((Nout + TILE - 1) / TILE, (M + TILE - 1) / TILE, batches);
  gemm_wmma_kernel<<<grid, 128, 0, st>>>(A, Bw, Cc, bias, resid, M, Nout, K,
                                         lda, ldb, ldc, a_so, a_si, b_so, b_si,
                                         c_so, c_si, batch_h, alpha, bT, act);
}

extern "C" void kernel_launch(void* const* d_in, const int* in_sizes, int n_in,
                              void* d_out, int out_size, void* d_ws, size_t ws_size,
                              hipStream_t stream) {
  (void)in_sizes; (void)n_in; (void)out_size; (void)ws_size;
  const float* x        = (const float*)d_in[0];
  const float* patch_w  = (const float*)d_in[1];
  const float* patch_b  = (const float*)d_in[2];
  const float* cls_tok  = (const float*)d_in[3];
  const float* pos_emb  = (const float*)d_in[4];
  const float* g_n1s    = (const float*)d_in[5];
  const float* g_n1b    = (const float*)d_in[6];
  const float* g_qk_w   = (const float*)d_in[7];
  const float* g_v_w    = (const float*)d_in[8];
  const float* g_proj_w = (const float*)d_in[9];
  const float* g_proj_b = (const float*)d_in[10];
  const float* g_pos_w  = (const float*)d_in[11];
  const float* g_pos_b  = (const float*)d_in[12];
  const float* g_gate   = (const float*)d_in[13];
  const float* g_n2s    = (const float*)d_in[14];
  const float* g_n2b    = (const float*)d_in[15];
  const float* g_fc1_w  = (const float*)d_in[16];
  const float* g_fc1_b  = (const float*)d_in[17];
  const float* g_fc2_w  = (const float*)d_in[18];
  const float* g_fc2_b  = (const float*)d_in[19];
  const float* m_n1s    = (const float*)d_in[20];
  const float* m_n1b    = (const float*)d_in[21];
  const float* m_qkv_w  = (const float*)d_in[22];
  const float* m_proj_w = (const float*)d_in[23];
  const float* m_proj_b = (const float*)d_in[24];
  const float* m_n2s    = (const float*)d_in[25];
  const float* m_n2b    = (const float*)d_in[26];
  const float* m_fc1_w  = (const float*)d_in[27];
  const float* m_fc1_b  = (const float*)d_in[28];
  const float* m_fc2_w  = (const float*)d_in[29];
  const float* m_fc2_b  = (const float*)d_in[30];
  const float* norm_s   = (const float*)d_in[31];
  const float* norm_b   = (const float*)d_in[32];
  const float* head_w   = (const float*)d_in[33];
  const float* head_b   = (const float*)d_in[34];
  float* out = (float*)d_out;
  float* ws  = (float*)d_ws;

  const int M196 = Bsz * N196;   // 12544
  const int M197 = Bsz * N197;   // 12608

  // workspace carve-up (float offsets); qkv region doubles as im2col buffer,
  // big region is shared by attention scores (padded rows) and MLP hidden.
  const long long o_qkv   = 0;                               // 29,048,832
  const long long o_f197  = o_qkv  + 29048832LL;             //  9,682,944
  const long long o_hln   = o_f197 + 9682944LL;              //  9,682,944
  const long long o_big   = o_hln  + 9682944LL;              // 38,731,776
  const long long o_obuf  = o_big  + 38731776LL;             //  9,682,944
  const long long o_pos   = o_obuf + 9682944LL;              //    460,992
  const long long o_f196  = o_pos  + 460992LL;               //  9,633,792
  const long long o_cls   = o_f196 + 9633792LL;              //     49,152

  float* qkv   = ws + o_qkv;
  float* f197  = ws + o_f197;
  float* hln   = ws + o_hln;
  float* big   = ws + o_big;          // scores OR mlp hidden
  float* obuf  = ws + o_obuf;
  float* posA  = ws + o_pos;
  float* f196  = ws + o_f196;
  float* clsln = ws + o_cls;
  float* vbuf  = qkv + (long long)M196 * C2;   // GPSA V lives behind QK

  // padded score strides
  const long long sg_si = (long long)N196 * SPAD;            // 40768
  const long long sg_so = (long long)NH * sg_si;             // 489216
  const long long sm_si = (long long)N197 * SPAD;            // 40976
  const long long sm_so = (long long)NH * sm_si;             // 491712

  // ---- patch embedding ----
  {
    int tot = M196 * Cdim;
    im2col_kernel<<<(tot + 255) / 256, 256, 0, stream>>>(x, qkv, tot);
    launch_gemm(stream, qkv, patch_w, f196, patch_b, nullptr,
                M196, Cdim, Cdim, Cdim, Cdim, Cdim,
                0, 0, 0, 0, 0, 0, 1, 1, 1.f, 0, 0);
    add_pos_kernel<<<(tot + 255) / 256, 256, 0, stream>>>(f196, pos_emb, tot);
  }

  // ---- GPSA blocks (tokens = 196) ----
  for (int i = 0; i < 3; ++i) {
    layernorm_kernel<<<M196, 256, 0, stream>>>(f196, hln,
        g_n1s + i * Cdim, g_n1b + i * Cdim, Cdim, Cdim);
    launch_gemm(stream, hln, g_qk_w + (long long)i * C2 * Cdim, qkv, nullptr, nullptr,
                M196, C2, Cdim, Cdim, Cdim, C2,
                0, 0, 0, 0, 0, 0, 1, 1, 1.f, 0, 0);
    launch_gemm(stream, hln, g_v_w + (long long)i * Cdim * Cdim, vbuf, nullptr, nullptr,
                M196, Cdim, Cdim, Cdim, Cdim, Cdim,
                0, 0, 0, 0, 0, 0, 1, 1, 1.f, 0, 0);
    {
      int tot = NH * N196 * N196;
      pos_logits_kernel<<<(tot + 255) / 256, 256, 0, stream>>>(
          g_pos_w + i * NH * 3, g_pos_b + i * NH, posA, tot);
      softmax_kernel<<<NH * N196, 256, 0, stream>>>(posA, N196, N196);
    }
    // S = scale * Q K^T per (b,h), padded rows
    launch_gemm(stream, qkv, qkv + Cdim, big, nullptr, nullptr,
                N196, N196, Dh, C2, C2, SPAD,
                (long long)N196 * C2, Dh, (long long)N196 * C2, Dh,
                sg_so, sg_si, Bsz * NH, NH, SCALE_, 0, 0);
    gpsa_blend_kernel<<<Bsz * NH * N196, 256, 0, stream>>>(big, posA, g_gate + i * NH);
    // O = attn @ V  (B transposed load), write into [b, n, h*64+d]
    launch_gemm(stream, big, vbuf, obuf, nullptr, nullptr,
                N196, Dh, N196, SPAD, Cdim, Cdim,
                sg_so, sg_si,
                (long long)N196 * Cdim, Dh,
                (long long)N196 * Cdim, Dh,
                Bsz * NH, NH, 1.f, 1, 0);
    launch_gemm(stream, obuf, g_proj_w + (long long)i * Cdim * Cdim, f196,
                g_proj_b + i * Cdim, f196,
                M196, Cdim, Cdim, Cdim, Cdim, Cdim,
                0, 0, 0, 0, 0, 0, 1, 1, 1.f, 0, 0);
    layernorm_kernel<<<M196, 256, 0, stream>>>(f196, hln,
        g_n2s + i * Cdim, g_n2b + i * Cdim, Cdim, Cdim);
    launch_gemm(stream, hln, g_fc1_w + (long long)i * C4 * Cdim, big,
                g_fc1_b + i * C4, nullptr,
                M196, C4, Cdim, Cdim, Cdim, C4,
                0, 0, 0, 0, 0, 0, 1, 1, 1.f, 0, 1);
    launch_gemm(stream, big, g_fc2_w + (long long)i * Cdim * C4, f196,
                g_fc2_b + i * Cdim, f196,
                M196, Cdim, C4, C4, C4, Cdim,
                0, 0, 0, 0, 0, 0, 1, 1, 1.f, 0, 0);
  }

  // ---- prepend cls token ----
  {
    int tot = M197 * Cdim;
    concat_cls_kernel<<<(tot + 255) / 256, 256, 0, stream>>>(f196, cls_tok, f197, tot);
  }

  // ---- MHSA blocks (tokens = 197) ----
  for (int i = 0; i < 9; ++i) {
    layernorm_kernel<<<M197, 256, 0, stream>>>(f197, hln,
        m_n1s + i * Cdim, m_n1b + i * Cdim, Cdim, Cdim);
    launch_gemm(stream, hln, m_qkv_w + (long long)i * C3 * Cdim, qkv, nullptr, nullptr,
                M197, C3, Cdim, Cdim, Cdim, C3,
                0, 0, 0, 0, 0, 0, 1, 1, 1.f, 0, 0);
    launch_gemm(stream, qkv, qkv + Cdim, big, nullptr, nullptr,
                N197, N197, Dh, C3, C3, SPAD,
                (long long)N197 * C3, Dh, (long long)N197 * C3, Dh,
                sm_so, sm_si, Bsz * NH, NH, SCALE_, 0, 0);
    softmax_kernel<<<Bsz * NH * N197, 256, 0, stream>>>(big, N197, SPAD);
    launch_gemm(stream, big, qkv + 2 * Cdim, obuf, nullptr, nullptr,
                N197, Dh, N197, SPAD, C3, Cdim,
                sm_so, sm_si,
                (long long)N197 * C3, Dh,
                (long long)N197 * Cdim, Dh,
                Bsz * NH, NH, 1.f, 1, 0);
    launch_gemm(stream, obuf, m_proj_w + (long long)i * Cdim * Cdim, f197,
                m_proj_b + i * Cdim, f197,
                M197, Cdim, Cdim, Cdim, Cdim, Cdim,
                0, 0, 0, 0, 0, 0, 1, 1, 1.f, 0, 0);
    layernorm_kernel<<<M197, 256, 0, stream>>>(f197, hln,
        m_n2s + i * Cdim, m_n2b + i * Cdim, Cdim, Cdim);
    launch_gemm(stream, hln, m_fc1_w + (long long)i * C4 * Cdim, big,
                m_fc1_b + i * C4, nullptr,
                M197, C4, Cdim, Cdim, Cdim, C4,
                0, 0, 0, 0, 0, 0, 1, 1, 1.f, 0, 1);
    launch_gemm(stream, big, m_fc2_w + (long long)i * Cdim * C4, f197,
                m_fc2_b + i * Cdim, f197,
                M197, Cdim, C4, C4, C4, Cdim,
                0, 0, 0, 0, 0, 0, 1, 1, 1.f, 0, 0);
  }

  // ---- final LN on cls rows + classifier head ----
  layernorm_kernel<<<Bsz, 256, 0, stream>>>(f197, clsln, norm_s, norm_b,
      (long long)N197 * Cdim, Cdim);
  launch_gemm(stream, clsln, head_w, out, head_b, nullptr,
              Bsz, NCLS, Cdim, Cdim, Cdim, NCLS,
              0, 0, 0, 0, 0, 0, 1, 1, 1.f, 0, 0);
}
